// DiFormer_44753559224983
// MI455X (gfx1250) — matrix operand order repaired and checked by reading, not compiled, not run on hardware
//
#include <hip/hip_runtime.h>
#include <hip/hip_bf16.h>
#include <math.h>

// ---------------------------------------------------------------------------
// Types for CDNA5 WMMA (wave32): v_wmma_f32_16x16x32_bf16
// ---------------------------------------------------------------------------
typedef __attribute__((ext_vector_type(16))) __bf16 v16bf;
typedef __attribute__((ext_vector_type(8)))  float  v8f;

#define DEV __device__ __forceinline__

DEV v16bf frag_load32B(const __bf16* p) {
    v16bf f;
    const uint4* s = reinterpret_cast<const uint4*>(p);
    uint4* d = reinterpret_cast<uint4*>(&f);
    d[0] = s[0]; d[1] = s[1];
    return f;
}

// A fragment (16x32 bf16, row-major source, lda in elements, multiple of 8):
// lanes 0-15 : row = base+lane,  elems K = k0+0..15
// lanes 16-31: row = base+lane-16, elems K = k0+16..31
DEV v16bf load_A_frag(const __bf16* A, int lda, int row, int k0, int lane) {
    const __bf16* p = A + (size_t)row * lda + k0 + ((lane >> 4) << 4);
    v16bf f;
    uint4* d = reinterpret_cast<uint4*>(&f);
    d[0] = *reinterpret_cast<const uint4*>(p);
    d[1] = *reinterpret_cast<const uint4*>(p + 8);
    return f;
}

DEV float gelu_tanh(float x) {
    return 0.5f * x * (1.0f + tanhf(0.7978845608f * (x + 0.044715f * x * x * x)));
}

// Async global -> LDS copy of one 16B chunk (per-lane), tracked by ASYNCcnt.
DEV void async_copy16(unsigned int lds_byte, const void* gsrc) {
    asm volatile("global_load_async_to_lds_b128 %0, %1, off"
                 :: "v"(lds_byte), "v"(gsrc) : "memory");
}
DEV void wait_async0() {
    asm volatile("s_wait_asynccnt 0x0" ::: "memory");
}

// ---------------------------------------------------------------------------
// Weight pack: f32 row-major (K x N) -> bf16 fragment-major
// layout: [kt][nt][lane(32)][e(16)], lanes 0-15 hold K=kt*32+e (col nt*16+lane),
// lanes 16-31 hold K=kt*32+16+e.   One tile = 1024 bytes, contiguous.
// ---------------------------------------------------------------------------
__global__ void pack_w_kernel(const float* __restrict__ W, __bf16* __restrict__ out,
                              int K, int N) {
    int tid = blockIdx.x * blockDim.x + threadIdx.x;
    int total = (K >> 5) * (N >> 4) * 32;
    if (tid >= total) return;
    int lane = tid & 31;
    int tile = tid >> 5;
    int NT = N >> 4;
    int nt = tile % NT;
    int kt = tile / NT;
    int n  = nt * 16 + (lane & 15);
    int kb = kt * 32 + ((lane >> 4) << 4);
    __bf16* o = out + (size_t)tid * 16;
    #pragma unroll
    for (int e = 0; e < 16; ++e)
        o[e] = (__bf16)W[(size_t)(kb + e) * N + n];
}

// ---------------------------------------------------------------------------
// Generic WMMA GEMM: C(MxN) = A(MxK,bf16 rowmajor) * Bpacked(KxN) [+bias]
// Block: 256 threads (8 wave32), 128x128 tile; wave = 32x64 (2x4 WMMA tiles).
// Packed-B tile for one K-step (8 n-tiles x 1KB) is staged into LDS via
// async global->LDS copies, double-buffered across K-steps.
// mode 0: Cf = val                       (f32)
// mode 1: Cb = bf16(gelu(val))
// mode 2: Xres += gate[n]*val            (in-place f32 residual)
// mode 3: Xres += gate[n]*(val+extra)
// mode 4: Cb = bf16(val)
// ---------------------------------------------------------------------------
__global__ void __launch_bounds__(256) gemm_wmma_kernel(
        const __bf16* __restrict__ A, int lda,
        const __bf16* __restrict__ Bp, int NT,
        const float* __restrict__ bias,
        float* __restrict__ Cf, __bf16* __restrict__ Cb, int ldc,
        float* __restrict__ Xres, int ldx,
        const float* __restrict__ gate,
        const float* __restrict__ extra, int lde,
        int M, int N, int K, int mode)
{
    __shared__ __align__(16) __bf16 Bs[2][4096];   // 2 x 8KB double buffer

    const int tid  = threadIdx.x;
    const int lane = tid & 31;
    const int wid  = tid >> 5;
    const int wm = wid & 3, wn = wid >> 2;
    const int m_base = blockIdx.y * 128 + wm * 32;
    const int n_base = blockIdx.x * 128 + wn * 64;
    const int blockNT0 = blockIdx.x * 8;

    v8f acc[2][4] = {};

    int r0 = m_base + (lane & 15);      if (r0 > M - 1) r0 = M - 1;
    int r1 = m_base + 16 + (lane & 15); if (r1 > M - 1) r1 = M - 1;

    // staging plan: 512 chunks of 16B per K-step; this thread copies 2 chunks.
    const unsigned int ldsBase = (unsigned int)(size_t)(&Bs[0][0]);
    const int ch0 = tid << 1, ch1 = ch0 + 1;
    int nt0 = blockNT0 + (ch0 >> 6); if (nt0 > NT - 1) nt0 = NT - 1;
    int nt1 = blockNT0 + (ch1 >> 6); if (nt1 > NT - 1) nt1 = NT - 1;
    const size_t sOff0 = (size_t)nt0 * 1024 + (size_t)(ch0 & 63) * 16;
    const size_t sOff1 = (size_t)nt1 * 1024 + (size_t)(ch1 & 63) * 16;
    const unsigned int dOff0 = ldsBase + (unsigned int)ch0 * 16;
    const unsigned int dOff1 = ldsBase + (unsigned int)ch1 * 16;
    const char* BpB = (const char*)Bp;

    const int KT = K >> 5;
    // prologue: stage kt=0 into buffer 0
    {
        const char* src = BpB + (size_t)0 * NT * 1024;
        async_copy16(dOff0, src + sOff0);
        async_copy16(dOff1, src + sOff1);
    }

    for (int kt = 0; kt < KT; ++kt) {
        wait_async0();        // my copies for step kt have landed in LDS
        __syncthreads();      // everyone's copies landed; prev buffer fully consumed
        if (kt + 1 < KT) {
            const int nb = (kt + 1) & 1;
            const char* src = BpB + (size_t)(kt + 1) * NT * 1024;
            async_copy16(dOff0 + nb * 8192u, src + sOff0);
            async_copy16(dOff1 + nb * 8192u, src + sOff1);
        }
        const int buf = kt & 1;
        const int k0 = kt << 5;
        v16bf a0 = load_A_frag(A, lda, r0, k0, lane);
        v16bf a1 = load_A_frag(A, lda, r1, k0, lane);
        #pragma unroll
        for (int j = 0; j < 4; ++j) {
            v16bf b = frag_load32B(&Bs[buf][(((wn << 2) + j) * 32 + lane) * 16]);
            acc[0][j] = __builtin_amdgcn_wmma_f32_16x16x32_bf16(false, a0, false, b, (short)0, acc[0][j], false, false);
            acc[1][j] = __builtin_amdgcn_wmma_f32_16x16x32_bf16(false, a1, false, b, (short)0, acc[1][j], false, false);
        }
    }

    #pragma unroll
    for (int i = 0; i < 2; ++i) {
        const int mrow = m_base + i * 16 + ((lane >> 4) << 3);
        #pragma unroll
        for (int j = 0; j < 4; ++j) {
            const int col = n_base + j * 16 + (lane & 15);
            if (col >= N) continue;
            const float bv = bias ? bias[col] : 0.0f;
            const float gv = (mode == 2 || mode == 3) ? gate[col] : 0.0f;
            #pragma unroll
            for (int r = 0; r < 8; ++r) {
                const int m = mrow + r;
                if (m >= M) break;
                float val = acc[i][j][r] + bv;
                if (mode == 0)      Cf[(size_t)m * ldc + col] = val;
                else if (mode == 1) Cb[(size_t)m * ldc + col] = (__bf16)gelu_tanh(val);
                else if (mode == 2) Xres[(size_t)m * ldx + col] += gv * val;
                else if (mode == 3) Xres[(size_t)m * ldx + col] += gv * (val + extra[(size_t)m * lde + col]);
                else                Cb[(size_t)m * ldc + col] = (__bf16)val;
            }
        }
    }
}

// ---------------------------------------------------------------------------
// Attention S = scale * Q @ K^T, per-head batched over blockIdx.z.
// Q,K: [head][T][64] bf16 row-major. B frag loads are contiguous per lane.
// ---------------------------------------------------------------------------
__global__ void __launch_bounds__(256) attn_qk_kernel(
        const __bf16* __restrict__ Qb, const __bf16* __restrict__ Kb,
        float* __restrict__ S, int T, float scale)
{
    const int head = blockIdx.z;
    const __bf16* Q  = Qb + (size_t)head * T * 64;
    const __bf16* Km = Kb + (size_t)head * T * 64;
    float* Sh = S + (size_t)head * T * T;

    const int lane = threadIdx.x & 31;
    const int wid  = threadIdx.x >> 5;
    const int wm = wid & 3, wn = wid >> 2;
    const int m_base = blockIdx.y * 128 + wm * 32;
    const int n_base = blockIdx.x * 128 + wn * 64;

    v8f acc[2][4] = {};
    int r0 = m_base + (lane & 15);      if (r0 > T - 1) r0 = T - 1;
    int r1 = m_base + 16 + (lane & 15); if (r1 > T - 1) r1 = T - 1;

    #pragma unroll
    for (int kt = 0; kt < 2; ++kt) {
        const int k0 = kt << 5;
        v16bf a0 = load_A_frag(Q, 64, r0, k0, lane);
        v16bf a1 = load_A_frag(Q, 64, r1, k0, lane);
        #pragma unroll
        for (int j = 0; j < 4; ++j) {
            int n = n_base + j * 16 + (lane & 15);
            if (n > T - 1) n = T - 1;
            const __bf16* bp = Km + (size_t)n * 64 + k0 + ((lane >> 4) << 4);
            v16bf b = frag_load32B(bp);
            acc[0][j] = __builtin_amdgcn_wmma_f32_16x16x32_bf16(false, a0, false, b, (short)0, acc[0][j], false, false);
            acc[1][j] = __builtin_amdgcn_wmma_f32_16x16x32_bf16(false, a1, false, b, (short)0, acc[1][j], false, false);
        }
    }
    #pragma unroll
    for (int i = 0; i < 2; ++i) {
        const int mrow = m_base + i * 16 + ((lane >> 4) << 3);
        #pragma unroll
        for (int j = 0; j < 4; ++j) {
            const int col = n_base + j * 16 + (lane & 15);
            if (col >= T) continue;
            #pragma unroll
            for (int r = 0; r < 8; ++r) {
                const int m = mrow + r;
                if (m >= T) break;
                Sh[(size_t)m * T + col] = acc[i][j][r] * scale;
            }
        }
    }
}

// ---------------------------------------------------------------------------
// O = P @ V  per head (blockIdx.z). P: [head][T][T] bf16; V packed frag-major
// per head ([T/32][4][32][16]); output bf16 into attnb[t][head*64+d] (ldo).
// ---------------------------------------------------------------------------
__global__ void __launch_bounds__(256) attn_pv_kernel(
        const __bf16* __restrict__ Pall, const __bf16* __restrict__ Vpall,
        __bf16* __restrict__ Ob, int T, int ldo)
{
    const int head = blockIdx.z;
    const __bf16* A  = Pall + (size_t)head * T * T;
    const __bf16* Bp = Vpall + (size_t)head * (size_t)(T / 32) * 4 * 512;
    __bf16* C = Ob + head * 64;

    const int lane = threadIdx.x & 31;
    const int wid  = threadIdx.x >> 5;
    const int wm = wid & 3, wn = wid >> 2;
    const int m_base = blockIdx.y * 128 + wm * 32;
    const int n_base = blockIdx.x * 128 + wn * 64;
    const int NT = 4, N = 64;

    v8f acc[2][4] = {};
    int r0 = m_base + (lane & 15);      if (r0 > T - 1) r0 = T - 1;
    int r1 = m_base + 16 + (lane & 15); if (r1 > T - 1) r1 = T - 1;

    int ntg[4];
    #pragma unroll
    for (int j = 0; j < 4; ++j) {
        int t = (n_base >> 4) + j;
        ntg[j] = (t < NT) ? t : (NT - 1);
    }

    const int KT = T >> 5;
    for (int kt = 0; kt < KT; ++kt) {
        const int k0 = kt << 5;
        v16bf a0 = load_A_frag(A, T, r0, k0, lane);
        v16bf a1 = load_A_frag(A, T, r1, k0, lane);
        #pragma unroll
        for (int j = 0; j < 4; ++j) {
            const __bf16* bp = Bp + ((((size_t)kt * NT + ntg[j]) << 5) + lane) * 16;
            v16bf b = frag_load32B(bp);
            acc[0][j] = __builtin_amdgcn_wmma_f32_16x16x32_bf16(false, a0, false, b, (short)0, acc[0][j], false, false);
            acc[1][j] = __builtin_amdgcn_wmma_f32_16x16x32_bf16(false, a1, false, b, (short)0, acc[1][j], false, false);
        }
    }
    #pragma unroll
    for (int i = 0; i < 2; ++i) {
        const int mrow = m_base + i * 16 + ((lane >> 4) << 3);
        #pragma unroll
        for (int j = 0; j < 4; ++j) {
            const int col = n_base + j * 16 + (lane & 15);
            if (col >= N) continue;
            #pragma unroll
            for (int r = 0; r < 8; ++r) {
                const int m = mrow + r;
                if (m >= T) break;
                C[(size_t)m * ldo + col] = (__bf16)acc[i][j][r];
            }
        }
    }
}

// ---------------------------------------------------------------------------
// Row softmax over f32 scores -> bf16 probs. grid=(T, H), block=256.
// ---------------------------------------------------------------------------
__global__ void softmax_kernel(const float* __restrict__ S, __bf16* __restrict__ P, int T) {
    const int head = blockIdx.y;
    const int row  = blockIdx.x;
    const float* sr = S + ((size_t)head * T + row) * T;
    __bf16* pr = P + ((size_t)head * T + row) * T;
    __shared__ float red[256];
    const int tid = threadIdx.x;

    float mx = -1e30f;
    for (int i = tid; i < T; i += 256) mx = fmaxf(mx, sr[i]);
    red[tid] = mx; __syncthreads();
    for (int st = 128; st > 0; st >>= 1) {
        if (tid < st) red[tid] = fmaxf(red[tid], red[tid + st]);
        __syncthreads();
    }
    mx = red[0]; __syncthreads();

    float sum = 0.f;
    for (int i = tid; i < T; i += 256) sum += expf(sr[i] - mx);
    red[tid] = sum; __syncthreads();
    for (int st = 128; st > 0; st >>= 1) {
        if (tid < st) red[tid] += red[tid + st];
        __syncthreads();
    }
    const float inv = 1.0f / red[0];
    for (int i = tid; i < T; i += 256)
        pr[i] = (__bf16)(expf(sr[i] - mx) * inv);
}

// ---------------------------------------------------------------------------
// LayerNorm + modulation:  out_bf16 = (1+scale[n])*LN(x) + shift[n]
// one block per row, hidden = 1024.
// ---------------------------------------------------------------------------
__global__ void ln_mod_kernel(const float* __restrict__ X, int ld,
                              const float* __restrict__ shift,
                              const float* __restrict__ scale,
                              __bf16* __restrict__ out, int ldo)
{
    const int row = blockIdx.x;
    const float* xr = X + (size_t)row * ld;
    __shared__ float r1[256], r2[256];
    const int tid = threadIdx.x;
    float s = 0.f, s2 = 0.f;
    for (int i = tid; i < 1024; i += 256) { float v = xr[i]; s += v; s2 += v * v; }
    r1[tid] = s; r2[tid] = s2; __syncthreads();
    for (int st = 128; st > 0; st >>= 1) {
        if (tid < st) { r1[tid] += r1[tid + st]; r2[tid] += r2[tid + st]; }
        __syncthreads();
    }
    const float mean = r1[0] * (1.0f / 1024.0f);
    const float var  = r2[0] * (1.0f / 1024.0f) - mean * mean;
    const float rstd = rsqrtf(var + 1e-6f);
    __bf16* orow = out + (size_t)row * ldo;
    for (int i = tid; i < 1024; i += 256) {
        float v = (xr[i] - mean) * rstd * (1.0f + scale[i]) + shift[i];
        orow[i] = (__bf16)v;
    }
}

// ---------------------------------------------------------------------------
// QKV prep: per (token, head) apply RMS-norm(q,k) with per-dim scale, RoPE,
// write q,k bf16 [head][T][64]; pack v into frag-major [head][kt][nt][lane][16].
// grid = (Tseg, 16), block = 64.
// ---------------------------------------------------------------------------
__global__ void prep_qkv_kernel(const float* __restrict__ QKV, int t_off,
                                const float* __restrict__ qs, const float* __restrict__ ks,
                                const float* __restrict__ pec, const float* __restrict__ pes,
                                __bf16* __restrict__ qb, __bf16* __restrict__ kb,
                                __bf16* __restrict__ vp, int T)
{
    const int t = blockIdx.x, head = blockIdx.y, d = threadIdx.x;
    const int tg = t + t_off;
    const float* base = QKV + (size_t)t * 3072 + head * 64;
    const float q = base[d];
    const float k = base[1024 + d];
    const float v = base[2048 + d];

    __shared__ float sq[64], sk[64];
    sq[d] = q * q; sk[d] = k * k; __syncthreads();
    for (int st = 32; st > 0; st >>= 1) {
        if (d < st) { sq[d] += sq[d + st]; sk[d] += sk[d + st]; }
        __syncthreads();
    }
    const float qr = rsqrtf(sq[0] * (1.0f / 64.0f) + 1e-6f);
    const float kr = rsqrtf(sk[0] * (1.0f / 64.0f) + 1e-6f);
    __syncthreads();
    const float qn = q * qr * qs[d];
    const float kn = k * kr * ks[d];
    sq[d] = qn; sk[d] = kn; __syncthreads();

    const int j = d >> 1;
    const float c = pec[(size_t)tg * 32 + j];
    const float s = pes[(size_t)tg * 32 + j];
    float qo, ko;
    if ((d & 1) == 0) { qo = c * sq[d] - s * sq[d + 1]; ko = c * sk[d] - s * sk[d + 1]; }
    else              { qo = s * sq[d - 1] + c * sq[d]; ko = s * sk[d - 1] + c * sk[d]; }

    const size_t qi = ((size_t)head * T + tg) * 64 + d;
    qb[qi] = (__bf16)qo;
    kb[qi] = (__bf16)ko;

    const int kt = tg >> 5, kk = tg & 31, nt = d >> 4;
    const int lane = (d & 15) + ((kk >= 16) ? 16 : 0);
    const int e = kk & 15;
    const size_t vi = (size_t)head * ((size_t)(T / 32) * 4 * 512) +
                      (((size_t)kt * 4 + nt) * 32 + lane) * 16 + e;
    vp[vi] = (__bf16)v;
}

// ---------------------------------------------------------------------------
// Small f32 GEMV: out[n] (+)= act(sum_k x[k]*W[k*N+n] + b[n]); act 1 = silu.
// ---------------------------------------------------------------------------
__global__ void gemv_f32_kernel(const float* __restrict__ x, const float* __restrict__ W,
                                const float* __restrict__ b, float* __restrict__ out,
                                int K, int N, int act, int accum)
{
    const int n = blockIdx.x * blockDim.x + threadIdx.x;
    if (n >= N) return;
    float s = 0.f;
    for (int k = 0; k < K; ++k) s += x[k] * W[(size_t)k * N + n];
    s += b[n];
    if (act == 1) s = s / (1.0f + expf(-s));
    if (accum) out[n] += s; else out[n] = s;
}

// timestep embedding (dim 256): [cos(t*1000*f_i) | sin(...)], f_i = 10000^(-i/128)
__global__ void temb_kernel(const float* __restrict__ t, float* __restrict__ out) {
    const int i = threadIdx.x;
    const float tv = t[0] * 1000.0f;
    if (i < 128) {
        float f = expf(-logf(10000.0f) * (float)i / 128.0f);
        out[i] = cosf(tv * f);
    } else {
        int j = i - 128;
        float f = expf(-logf(10000.0f) * (float)j / 128.0f);
        out[i] = sinf(tv * f);
    }
}

__global__ void silu_kernel(const float* __restrict__ in, float* __restrict__ out, int n) {
    const int i = blockIdx.x * blockDim.x + threadIdx.x;
    if (i < n) { float v = in[i]; out[i] = v / (1.0f + expf(-v)); }
}

__global__ void f2bf_kernel(const float* __restrict__ in, __bf16* __restrict__ out, int n) {
    const int i = blockIdx.x * blockDim.x + threadIdx.x;
    if (i < n) out[i] = (__bf16)in[i];
}

// RoPE cos/sin tables: [1280][32]; axes dims (16,24,24) -> (8,12,12) pairs.
__global__ void pe_kernel(const int* __restrict__ txt_ids, const int* __restrict__ img_ids,
                          float* __restrict__ pc, float* __restrict__ ps)
{
    const int idx = blockIdx.x * blockDim.x + threadIdx.x;
    if (idx >= 1280 * 32) return;
    const int t = idx >> 5, j = idx & 31;
    int axis, i0, dim;
    if (j < 8)       { axis = 0; i0 = j;      dim = 16; }
    else if (j < 20) { axis = 1; i0 = j - 8;  dim = 24; }
    else             { axis = 2; i0 = j - 20; dim = 24; }
    const int id = (t < 256) ? txt_ids[t * 3 + axis] : img_ids[(t - 256) * 3 + axis];
    const float omega = powf(10000.0f, -2.0f * (float)i0 / (float)dim);
    const float arg = (float)id * omega;
    pc[idx] = cosf(arg);
    ps[idx] = sinf(arg);
}

// ---------------------------------------------------------------------------
// Host orchestration
// ---------------------------------------------------------------------------
extern "C" void kernel_launch(void* const* d_in, const int* in_sizes, int n_in,
                              void* d_out, int out_size, void* d_ws, size_t ws_size,
                              hipStream_t stream) {
    // ---- inputs (setup_inputs order) ----
    const float* img        = (const float*)d_in[0];
    const float* txt        = (const float*)d_in[1];
    const float* t_in       = (const float*)d_in[2];
    const float* guid       = (const float*)d_in[3];
    const float* y_in       = (const float*)d_in[4];
    const int*   img_ids    = (const int*)d_in[5];
    const int*   txt_ids    = (const int*)d_in[6];
    const float* img_in_w   = (const float*)d_in[7];
    const float* img_in_b   = (const float*)d_in[8];
    const float* txt_in_w   = (const float*)d_in[9];
    const float* txt_in_b   = (const float*)d_in[10];
    const float* t_w1 = (const float*)d_in[11];  const float* t_b1 = (const float*)d_in[12];
    const float* t_w2 = (const float*)d_in[13];  const float* t_b2 = (const float*)d_in[14];
    const float* g_w1 = (const float*)d_in[15];  const float* g_b1 = (const float*)d_in[16];
    const float* g_w2 = (const float*)d_in[17];  const float* g_b2 = (const float*)d_in[18];
    const float* v_w1 = (const float*)d_in[19];  const float* v_b1 = (const float*)d_in[20];
    const float* v_w2 = (const float*)d_in[21];  const float* v_b2 = (const float*)d_in[22];
    const float* db_img_mod_w = (const float*)d_in[23]; const float* db_img_mod_b = (const float*)d_in[24];
    const float* db_txt_mod_w = (const float*)d_in[25]; const float* db_txt_mod_b = (const float*)d_in[26];
    const float* db_img_qkv_w = (const float*)d_in[27]; const float* db_img_qkv_b = (const float*)d_in[28];
    const float* db_img_qs    = (const float*)d_in[29]; const float* db_img_ks    = (const float*)d_in[30];
    const float* db_img_o_w   = (const float*)d_in[31]; const float* db_img_o_b   = (const float*)d_in[32];
    const float* db_img_m1_w  = (const float*)d_in[33]; const float* db_img_m1_b  = (const float*)d_in[34];
    const float* db_img_m2_w  = (const float*)d_in[35]; const float* db_img_m2_b  = (const float*)d_in[36];
    const float* db_txt_qkv_w = (const float*)d_in[37]; const float* db_txt_qkv_b = (const float*)d_in[38];
    const float* db_txt_qs    = (const float*)d_in[39]; const float* db_txt_ks    = (const float*)d_in[40];
    const float* db_txt_o_w   = (const float*)d_in[41]; const float* db_txt_o_b   = (const float*)d_in[42];
    const float* db_txt_m1_w  = (const float*)d_in[43]; const float* db_txt_m1_b  = (const float*)d_in[44];
    const float* db_txt_m2_w  = (const float*)d_in[45]; const float* db_txt_m2_b  = (const float*)d_in[46];
    const float* sb_mod_w = (const float*)d_in[47]; const float* sb_mod_b = (const float*)d_in[48];
    const float* sb_qkv_w = (const float*)d_in[49]; const float* sb_qkv_b = (const float*)d_in[50];
    const float* sb_qs    = (const float*)d_in[51]; const float* sb_ks    = (const float*)d_in[52];
    const float* sb_o_w   = (const float*)d_in[53]; const float* sb_o_b   = (const float*)d_in[54];
    const float* sb_m1_w  = (const float*)d_in[55]; const float* sb_m1_b  = (const float*)d_in[56];
    const float* sb_m2_w  = (const float*)d_in[57]; const float* sb_m2_b  = (const float*)d_in[58];
    const float* fl_mod_w = (const float*)d_in[59]; const float* fl_mod_b = (const float*)d_in[60];
    const float* fl_w     = (const float*)d_in[61]; const float* fl_b     = (const float*)d_in[62];

    (void)in_sizes; (void)n_in; (void)out_size; (void)ws_size;

    const int T = 1280, TT = 256, TI = 1024, H = 1024, NH = 16;

    // ---- workspace bump allocator ----
    char* wsb = (char*)d_ws;
    size_t off = 0;
    auto alloc = [&](size_t bytes) -> void* {
        void* p = wsb + off;
        off += (bytes + 255) & ~(size_t)255;
        return p;
    };
    float*  temb = (float*)alloc(256 * 4);
    float*  gemb = (float*)alloc(256 * 4);
    float*  hid  = (float*)alloc(1024 * 4);
    float*  vec  = (float*)alloc(1024 * 4);
    float*  svec = (float*)alloc(1024 * 4);
    float*  pec  = (float*)alloc((size_t)T * 32 * 4);
    float*  pes  = (float*)alloc((size_t)T * 32 * 4);
    float*  modI = (float*)alloc(6144 * 4);
    float*  modT = (float*)alloc(6144 * 4);
    float*  X    = (float*)alloc((size_t)T * H * 4);
    __bf16* Xm   = (__bf16*)alloc((size_t)T * H * 2);
    float*  QKV  = (float*)alloc((size_t)T * 3072 * 4);
    __bf16* qb   = (__bf16*)alloc((size_t)NH * T * 64 * 2);
    __bf16* kb   = (__bf16*)alloc((size_t)NH * T * 64 * 2);
    __bf16* vp   = (__bf16*)alloc((size_t)NH * (T / 32) * 4 * 512 * 2);
    float*  Sb   = (float*)alloc((size_t)NH * T * T * 4);
    __bf16* Pb   = (__bf16*)alloc((size_t)NH * T * T * 2);
    __bf16* attnb= (__bf16*)alloc((size_t)T * H * 2);
    __bf16* Hb   = (__bf16*)alloc((size_t)T * 4096 * 2);
    float*  Tmp  = (float*)alloc((size_t)T * H * 4);
    __bf16* Wp   = (__bf16*)alloc((size_t)4096 * 1024 * 2);
    __bf16* inb  = (__bf16*)alloc((size_t)TT * 768 * 2);

    // ---- host helpers ----
    auto pack = [&](const float* W, int K, int N) {
        int total = (K / 32) * (N / 16) * 32;
        pack_w_kernel<<<(total + 255) / 256, 256, 0, stream>>>(W, Wp, K, N);
    };
    auto gemm = [&](const __bf16* A, int lda, int NT, const float* bias,
                    float* Cf, __bf16* Cb, int ldc,
                    float* Xres, int ldx, const float* gate,
                    const float* extra, int lde, int M, int N, int K, int mode) {
        dim3 grid((N + 127) / 128, (M + 127) / 128);
        gemm_wmma_kernel<<<grid, 256, 0, stream>>>(A, lda, Wp, NT, bias, Cf, Cb, ldc,
                                                   Xres, ldx, gate, extra, lde, M, N, K, mode);
    };
    auto gemv = [&](const float* x, const float* W, const float* b, float* out,
                    int K, int N, int act, int accum) {
        gemv_f32_kernel<<<(N + 255) / 256, 256, 0, stream>>>(x, W, b, out, K, N, act, accum);
    };
    auto ln = [&](const float* Xp, const float* shift, const float* scale,
                  __bf16* out, int rows) {
        ln_mod_kernel<<<rows, 256, 0, stream>>>(Xp, H, shift, scale, out, H);
    };
    auto prep = [&](const float* qkvp, int Tseg, int toff, const float* qs, const float* ks) {
        dim3 g(Tseg, NH);
        prep_qkv_kernel<<<g, 64, 0, stream>>>(qkvp, toff, qs, ks, pec, pes, qb, kb, vp, T);
    };
    auto attention = [&]() {
        dim3 gqk(T / 128, T / 128, NH);
        attn_qk_kernel<<<gqk, 256, 0, stream>>>(qb, kb, Sb, T, 0.125f);
        dim3 gsm(T, NH);
        softmax_kernel<<<gsm, 256, 0, stream>>>(Sb, Pb, T);
        dim3 gpv(1, T / 128, NH);
        attn_pv_kernel<<<gpv, 256, 0, stream>>>(Pb, vp, attnb, T, H);
    };

    // ---- vec / svec (timestep + guidance + y embedders) ----
    temb_kernel<<<1, 256, 0, stream>>>(t_in, temb);
    temb_kernel<<<1, 256, 0, stream>>>(guid, gemb);
    gemv(temb, t_w1, t_b1, hid, 256, 1024, 1, 0);
    gemv(hid,  t_w2, t_b2, vec, 1024, 1024, 0, 0);
    gemv(gemb, g_w1, g_b1, hid, 256, 1024, 1, 0);
    gemv(hid,  g_w2, g_b2, vec, 1024, 1024, 0, 1);
    gemv(y_in, v_w1, v_b1, hid, 768, 1024, 1, 0);
    gemv(hid,  v_w2, v_b2, vec, 1024, 1024, 0, 1);
    silu_kernel<<<4, 256, 0, stream>>>(vec, svec, 1024);

    // ---- RoPE tables ----
    pe_kernel<<<(T * 32 + 255) / 256, 256, 0, stream>>>(txt_ids, img_ids, pec, pes);

    // ---- input projections: X = [txt_emb ; img_emb] ----
    f2bf_kernel<<<(TT * 768 + 255) / 256, 256, 0, stream>>>(txt, inb, TT * 768);
    pack(txt_in_w, 768, 1024);
    gemm(inb, 768, 64, txt_in_b, X, nullptr, H, nullptr, 0, nullptr, nullptr, 0, TT, 1024, 768, 0);
    f2bf_kernel<<<(TI * 64 + 255) / 256, 256, 0, stream>>>(img, inb, TI * 64);
    pack(img_in_w, 64, 1024);
    gemm(inb, 64, 64, img_in_b, X + (size_t)TT * H, nullptr, H, nullptr, 0, nullptr, nullptr, 0, TI, 1024, 64, 0);

    // ---- double-stream blocks ----
    for (int l = 0; l < 2; ++l) {
        const size_t MW = (size_t)1024 * 6144, QW = (size_t)1024 * 3072;
        const size_t OW = (size_t)1024 * 1024, M1 = (size_t)1024 * 4096, M2 = (size_t)4096 * 1024;
        gemv(svec, db_img_mod_w + l * MW, db_img_mod_b + l * 6144, modI, 1024, 6144, 0, 0);
        gemv(svec, db_txt_mod_w + l * MW, db_txt_mod_b + l * 6144, modT, 1024, 6144, 0, 0);

        ln(X, modT, modT + 1024, Xm, TT);
        ln(X + (size_t)TT * H, modI, modI + 1024, Xm + (size_t)TT * H, TI);

        pack(db_txt_qkv_w + l * QW, 1024, 3072);
        gemm(Xm, H, 192, db_txt_qkv_b + l * 3072, QKV, nullptr, 3072, nullptr, 0, nullptr, nullptr, 0, TT, 3072, 1024, 0);
        pack(db_img_qkv_w + l * QW, 1024, 3072);
        gemm(Xm + (size_t)TT * H, H, 192, db_img_qkv_b + l * 3072, QKV + (size_t)TT * 3072, nullptr, 3072, nullptr, 0, nullptr, nullptr, 0, TI, 3072, 1024, 0);

        prep(QKV, TT, 0, db_txt_qs + l * 64, db_txt_ks + l * 64);
        prep(QKV + (size_t)TT * 3072, TI, TT, db_img_qs + l * 64, db_img_ks + l * 64);
        attention();

        pack(db_txt_o_w + l * OW, 1024, 1024);
        gemm(attnb, H, 64, db_txt_o_b + l * 1024, nullptr, nullptr, 0, X, H, modT + 2048, nullptr, 0, TT, 1024, 1024, 2);
        pack(db_img_o_w + l * OW, 1024, 1024);
        gemm(attnb + (size_t)TT * H, H, 64, db_img_o_b + l * 1024, nullptr, nullptr, 0, X + (size_t)TT * H, H, modI + 2048, nullptr, 0, TI, 1024, 1024, 2);

        // txt MLP
        ln(X, modT + 3072, modT + 4096, Xm, TT);
        pack(db_txt_m1_w + l * M1, 1024, 4096);
        gemm(Xm, H, 256, db_txt_m1_b + l * 4096, nullptr, Hb, 4096, nullptr, 0, nullptr, nullptr, 0, TT, 4096, 1024, 1);
        pack(db_txt_m2_w + l * M2, 4096, 1024);
        gemm(Hb, 4096, 64, db_txt_m2_b + l * 1024, nullptr, nullptr, 0, X, H, modT + 5120, nullptr, 0, TT, 1024, 4096, 2);
        // img MLP
        ln(X + (size_t)TT * H, modI + 3072, modI + 4096, Xm, TI);
        pack(db_img_m1_w + l * M1, 1024, 4096);
        gemm(Xm, H, 256, db_img_m1_b + l * 4096, nullptr, Hb, 4096, nullptr, 0, nullptr, nullptr, 0, TI, 4096, 1024, 1);
        pack(db_img_m2_w + l * M2, 4096, 1024);
        gemm(Hb, 4096, 64, db_img_m2_b + l * 1024, nullptr, nullptr, 0, X + (size_t)TT * H, H, modI + 5120, nullptr, 0, TI, 1024, 4096, 2);
    }

    // ---- single-stream blocks ----
    for (int l = 0; l < 4; ++l) {
        const size_t MW = (size_t)1024 * 3072, QW = (size_t)1024 * 3072;
        const size_t OW = (size_t)1024 * 1024, M1 = (size_t)1024 * 4096, M2 = (size_t)4096 * 1024;
        gemv(svec, sb_mod_w + l * MW, sb_mod_b + l * 3072, modI, 1024, 3072, 0, 0);
        ln(X, modI, modI + 1024, Xm, T);

        pack(sb_qkv_w + l * QW, 1024, 3072);
        gemm(Xm, H, 192, sb_qkv_b + l * 3072, QKV, nullptr, 3072, nullptr, 0, nullptr, nullptr, 0, T, 3072, 1024, 0);
        prep(QKV, T, 0, sb_qs + l * 64, sb_ks + l * 64);
        attention();

        pack(sb_o_w + l * OW, 1024, 1024);
        gemm(attnb, H, 64, sb_o_b + l * 1024, Tmp, nullptr, H, nullptr, 0, nullptr, nullptr, 0, T, 1024, 1024, 0);
        pack(sb_m1_w + l * M1, 1024, 4096);
        gemm(Xm, H, 256, sb_m1_b + l * 4096, nullptr, Hb, 4096, nullptr, 0, nullptr, nullptr, 0, T, 4096, 1024, 1);
        pack(sb_m2_w + l * M2, 4096, 1024);
        gemm(Hb, 4096, 64, sb_m2_b + l * 1024, nullptr, nullptr, 0, X, H, modI + 2048, Tmp, H, T, 1024, 4096, 3);
    }

    // ---- final layer ----
    gemv(svec, fl_mod_w, fl_mod_b, modI, 1024, 2048, 0, 0);
    ln(X + (size_t)TT * H, modI, modI + 1024, Xm, TI);   // shift=modI[0:1024], scale=modI[1024:2048]
    pack(fl_w, 1024, 64);
    gemm(Xm, H, 4, fl_b, (float*)d_out, nullptr, 64, nullptr, 0, nullptr, nullptr, 0, TI, 64, 1024, 0);
}
